// TypeAreaMultiHeadAttention_34497177321780
// MI455X (gfx1250) — compile-verified, matrix operand order checked by reading, not run on hardware
//
#include <hip/hip_runtime.h>

// ---------------------------------------------------------------------------
// Problem constants: B=4, S=2048, D=1024, H=16, DK=DV=64
// ---------------------------------------------------------------------------
#define BB   4
#define SS   2048
#define DD   1024
#define HH   16
#define DH   64
#define NTOK (BB * SS)
#define INV_TEMP 0.0625f        // 1/sqrt(4*64)

typedef __attribute__((ext_vector_type(16))) __bf16 v16bf;
typedef __attribute__((ext_vector_type(8)))  __bf16 v8bf;
typedef __attribute__((ext_vector_type(2)))  __bf16 v2bf;
typedef __attribute__((ext_vector_type(8)))  float  v8f;

// ---------------------------------------------------------------------------
// Fragment-order storage (see round 1):
//   A (16x32, (m,k)): lane=(m&15)+16*((k>>3)&1), j=((k&31)>>4)*8+(k&7)
//   B (32x16, (k,n)): lane=(n&15)+16*((k>>3)&1), j=((k&31)>>4)*8+(k&7)
// Region = 512 bf16 = [lane][16]; fragment read = 32B contiguous per lane.
// ---------------------------------------------------------------------------
__device__ inline v16bf frag_ld(const __bf16* region, int lane) {
  return *reinterpret_cast<const v16bf*>(region + lane * 16);
}

__device__ inline v8f wmma_bf16(v16bf a, v16bf b, v8f c) {
  return __builtin_amdgcn_wmma_f32_16x16x32_bf16(
      false, a, false, b, (short)0, c, false, false);
}

__device__ inline v8f vzero8() {
  v8f z = {0.f, 0.f, 0.f, 0.f, 0.f, 0.f, 0.f, 0.f};
  return z;
}

// 16B async copy global -> LDS, tracked on ASYNCcnt; data bypasses VGPRs.
__device__ inline void async_ld_b128(const __bf16* lds_dst, const __bf16* gsrc) {
  const unsigned lds_off = (unsigned)(uintptr_t)(void*)lds_dst;
  const unsigned long long gaddr = (unsigned long long)(uintptr_t)gsrc;
  asm volatile("global_load_async_to_lds_b128 %0, %1, off"
               :: "v"(lds_off), "v"(gaddr)
               : "memory");
}
__device__ inline void wait_asynccnt0() {
  asm volatile("s_wait_asynccnt 0x0" ::: "memory");
}

// ---------------------------------------------------------------------------
// Kernel 0: weights fp32 [1024x1024] -> bf16 in B-fragment order
//   dst layout: [kt(32)][nt(64)][lane(32)][j(16)]
// ---------------------------------------------------------------------------
__global__ __launch_bounds__(256) void cvt_swizzle_kernel(
    const float* __restrict__ src, __bf16* __restrict__ dst) {
  const int g = blockIdx.x * blockDim.x + threadIdx.x;   // 0..131071
  const int jb   = (g & 1) * 8;
  const int lane = (g >> 1) & 31;
  const int nt   = (g >> 6) & 63;
  const int kt   = (g >> 12) & 31;
  const int n     = nt * 16 + (lane & 15);
  const int khalf = (lane >> 4) * 8;
  v8bf v;
#pragma unroll
  for (int jj = 0; jj < 8; ++jj) {
    const int j = jb + jj;
    const int k = kt * 32 + (j >> 3) * 16 + khalf + (j & 7);
    v[jj] = (__bf16)src[(size_t)k * DD + n];
  }
  *reinterpret_cast<v8bf*>(&dst[(size_t)g * 8]) = v;
}

// ---------------------------------------------------------------------------
// Kernel 1: fused QKV projection, out = x_a @ W_a + x_s @ W_s
// grid (NTOK/64, H, 3), block 128. Out: [B,H,S,64] bf16.
// ---------------------------------------------------------------------------
__global__ __launch_bounds__(128) void qkv_proj_kernel(
    const float* __restrict__ q_a, const float* __restrict__ q_s,
    const float* __restrict__ k_a, const float* __restrict__ k_s,
    const float* __restrict__ v_a, const float* __restrict__ v_s,
    const __bf16* __restrict__ Wqa, const __bf16* __restrict__ Wqs,
    const __bf16* __restrict__ Wka, const __bf16* __restrict__ Wks,
    const __bf16* __restrict__ Wva, const __bf16* __restrict__ Wvs,
    __bf16* __restrict__ Qw, __bf16* __restrict__ Kw, __bf16* __restrict__ Vw) {
  const float  *xs[2];
  const __bf16 *wp[2];
  __bf16*       out;
  if (blockIdx.z == 0)      { xs[0] = q_a; xs[1] = q_s; wp[0] = Wqa; wp[1] = Wqs; out = Qw; }
  else if (blockIdx.z == 1) { xs[0] = k_a; xs[1] = k_s; wp[0] = Wka; wp[1] = Wks; out = Kw; }
  else                      { xs[0] = v_a; xs[1] = v_s; wp[0] = Wva; wp[1] = Wvs; out = Vw; }

  const int row0 = blockIdx.x * 64;
  const int h    = blockIdx.y;
  const int tid  = threadIdx.x;
  const int wave = tid >> 5, lane = tid & 31;

  __shared__ __bf16 As[4 * 512];  // [mt][lane][j]

  v8f acc[4];
#pragma unroll
  for (int j = 0; j < 4; ++j) acc[j] = vzero8();

  for (int ph = 0; ph < 2; ++ph) {
    const float*  x = xs[ph];
    const __bf16* w = wp[ph];
    for (int kt = 0; kt < 32; ++kt) {
      __syncthreads();
      // stage A: 256 groups (8 k each), fp32->bf16, one b128 store per group
#pragma unroll
      for (int i = 0; i < 2; ++i) {
        const int g = tid + i * 128;
        const int r = g >> 2, c = g & 3;
        const float* p = &x[(size_t)(row0 + r) * DD + kt * 32 + c * 8];
        v8bf v;
#pragma unroll
        for (int jj = 0; jj < 8; ++jj) v[jj] = (__bf16)p[jj];
        *reinterpret_cast<v8bf*>(
            &As[(r >> 4) * 512 + ((r & 15) + 16 * (c & 1)) * 16 + (c >> 1) * 8]) = v;
      }
      __syncthreads();
      // issue all operand loads, then 4 back-to-back WMMAs
      const __bf16* wb = &w[((size_t)kt * 64 + h * 4) * 512];
      const v16bf b0 = frag_ld(wb + 0 * 512, lane);
      const v16bf b1 = frag_ld(wb + 1 * 512, lane);
      const v16bf b2 = frag_ld(wb + 2 * 512, lane);
      const v16bf b3 = frag_ld(wb + 3 * 512, lane);
      const v16bf a  = frag_ld(&As[wave * 512], lane);
      acc[0] = wmma_bf16(a, b0, acc[0]);
      acc[1] = wmma_bf16(a, b1, acc[1]);
      acc[2] = wmma_bf16(a, b2, acc[2]);
      acc[3] = wmma_bf16(a, b3, acc[3]);
    }
  }

  const int n = lane & 15, half = lane >> 4;
#pragma unroll
  for (int r = 0; r < 8; ++r) {
    const int t = row0 + wave * 16 + r + 8 * half;
    const int b = t / SS, s = t & (SS - 1);
#pragma unroll
    for (int j = 0; j < 4; ++j) {
      out[(((size_t)(b * HH + h) * SS) + s) * DH + j * 16 + n] = (__bf16)acc[j][r];
    }
  }
}

// ---------------------------------------------------------------------------
// Kernel 2: flash attention; grid (S/64, H, B), block 128.
// Q/K tiles staged with GLOBAL_LOAD_ASYNC_TO_LDS_B128 (ASYNCcnt path).
// ---------------------------------------------------------------------------
__global__ __launch_bounds__(128) void attn_kernel(
    const __bf16* __restrict__ Qw, const __bf16* __restrict__ Kw,
    const __bf16* __restrict__ Vw, __bf16* __restrict__ ctx) {
  const int qt  = blockIdx.x;
  const int h   = blockIdx.y;
  const int b   = blockIdx.z;
  const int tid = threadIdx.x;
  const int wave = tid >> 5, lane = tid & 31;
  const int m0   = wave * 16;

  const __bf16* Qh = Qw + (size_t)(b * HH + h) * SS * DH;
  const __bf16* Kh = Kw + (size_t)(b * HH + h) * SS * DH;
  const __bf16* Vh = Vw + (size_t)(b * HH + h) * SS * DH;

  __shared__ __bf16 Qa[8 * 512];  // A frags: [mt(4)][kt_d(2)]
  __shared__ __bf16 Kb[8 * 512];  // B frags (K^T): [kt_d(2)][nt_key(4)]
  __shared__ __bf16 Vb[8 * 512];  // B frags (V):   [kt_key(2)][nt_d(4)]
  __shared__ __bf16 Pa[8 * 512];  // A frags (P):   [mt(4)][kt_key(2)]

  // ---- stage Q via async copies: 512 x 16B groups -------------------------
#pragma unroll
  for (int i = 0; i < 4; ++i) {
    const int g = tid + i * 128;
    const int r = g >> 3, c = g & 7;
    const int kt = c >> 2, cc = c & 3;
    async_ld_b128(
        &Qa[((r >> 4) * 2 + kt) * 512 + ((r & 15) + 16 * (cc & 1)) * 16 + (cc >> 1) * 8],
        &Qh[(size_t)(qt * 64 + r) * DH + c * 8]);
  }
  wait_asynccnt0();
  __syncthreads();
  const v16bf aq0 = frag_ld(&Qa[(wave * 2 + 0) * 512], lane);
  const v16bf aq1 = frag_ld(&Qa[(wave * 2 + 1) * 512], lane);

  v8f cacc[4];
#pragma unroll
  for (int j = 0; j < 4; ++j) cacc[j] = vzero8();
  float mrow[8], lrow[8];
#pragma unroll
  for (int r = 0; r < 8; ++r) { mrow[r] = -1e30f; lrow[r] = 0.f; }

  for (int kt0 = 0; kt0 < SS / 64; ++kt0) {
    __syncthreads();
    // ---- K tile: async global->LDS, fragment-order scatter ----------------
#pragma unroll
    for (int i = 0; i < 4; ++i) {
      const int g = tid + i * 128;
      const int key = g >> 3, c = g & 7;
      const int dt = c >> 2, cc = c & 3;
      async_ld_b128(
          &Kb[(dt * 4 + (key >> 4)) * 512 + ((key & 15) + 16 * (cc & 1)) * 16 + (cc >> 1) * 8],
          &Kh[(size_t)(kt0 * 64 + key) * DH + c * 8]);
    }
    // ---- V tile: (key,key+1) pair interleave, overlaps with async K -------
#pragma unroll
    for (int i = 0; i < 2; ++i) {
      const int g = tid + i * 128;
      const int d0 = (g & 7) * 8;
      const int key = (g >> 3) * 2;
      const v8bf r0 = *reinterpret_cast<const v8bf*>(
          &Vh[(size_t)(kt0 * 64 + key) * DH + d0]);
      const v8bf r1 = *reinterpret_cast<const v8bf*>(
          &Vh[(size_t)(kt0 * 64 + key + 1) * DH + d0]);
      const int j    = ((key & 31) >> 4) * 8 + (key & 7);  // even
      const int half = (key >> 3) & 1;
      const int base = ((key >> 5) * 4 + (d0 >> 4)) * 512;
#pragma unroll
      for (int t = 0; t < 8; ++t) {
        v2bf v;
        v[0] = r0[t];
        v[1] = r1[t];
        *reinterpret_cast<v2bf*>(
            &Vb[base + (((d0 + t) & 15) + 16 * half) * 16 + j]) = v;
      }
    }
    wait_asynccnt0();
    __syncthreads();

    // ---- scores: preload all 8 K^T fragments, then 8 WMMAs ----------------
    v16bf kb[8];
#pragma unroll
    for (int j = 0; j < 4; ++j) {
      kb[j]     = frag_ld(&Kb[(0 * 4 + j) * 512], lane);
      kb[4 + j] = frag_ld(&Kb[(1 * 4 + j) * 512], lane);
    }
    v8f sacc[4];
#pragma unroll
    for (int j = 0; j < 4; ++j) {
      v8f z = vzero8();
      z = wmma_bf16(aq0, kb[j], z);
      z = wmma_bf16(aq1, kb[4 + j], z);
      sacc[j] = z;
    }

    // ---- online softmax on C-fragments ------------------------------------
#pragma unroll
    for (int r = 0; r < 8; ++r) {
      float tm = -1e30f;
#pragma unroll
      for (int j = 0; j < 4; ++j) {
        sacc[j][r] *= INV_TEMP;
        tm = fmaxf(tm, sacc[j][r]);
      }
      for (int d = 1; d < 16; d <<= 1) tm = fmaxf(tm, __shfl_xor(tm, d, 32));
      const float mnew = fmaxf(mrow[r], tm);
      const float corr = __expf(mrow[r] - mnew);
      float psum = 0.f;
#pragma unroll
      for (int j = 0; j < 4; ++j) {
        const float p = __expf(sacc[j][r] - mnew);
        sacc[j][r] = p;
        psum += p;
      }
      for (int d = 1; d < 16; d <<= 1) psum += __shfl_xor(psum, d, 32);
      lrow[r] = lrow[r] * corr + psum;
      mrow[r] = mnew;
#pragma unroll
      for (int j = 0; j < 4; ++j) cacc[j][r] *= corr;
    }

    // ---- P: C-layout registers -> A-fragment order (wave-private rows) ----
    {
      const int nl = lane & 15, ch = lane >> 4;
      const int thalf = (nl >> 3) & 1;
#pragma unroll
      for (int r = 0; r < 8; ++r) {
        const int m = m0 + r + 8 * ch;
        const int tlane = (m & 15) + 16 * thalf;
#pragma unroll
        for (int j16 = 0; j16 < 4; ++j16) {
          const int tj = (j16 & 1) * 8 + (nl & 7);
          Pa[(wave * 2 + (j16 >> 1)) * 512 + tlane * 16 + tj] =
              (__bf16)sacc[j16][r];
        }
      }
    }
    // same-wave DS store->load stays in order; no barrier needed
    const v16bf ap0 = frag_ld(&Pa[(wave * 2 + 0) * 512], lane);
    const v16bf ap1 = frag_ld(&Pa[(wave * 2 + 1) * 512], lane);

    // ---- ctx += P @ V: preload all 8 V fragments, then 8 WMMAs ------------
    v16bf vb[8];
#pragma unroll
    for (int j = 0; j < 4; ++j) {
      vb[j]     = frag_ld(&Vb[(0 * 4 + j) * 512], lane);
      vb[4 + j] = frag_ld(&Vb[(1 * 4 + j) * 512], lane);
    }
#pragma unroll
    for (int j = 0; j < 4; ++j) {
      cacc[j] = wmma_bf16(ap0, vb[j], cacc[j]);
      cacc[j] = wmma_bf16(ap1, vb[4 + j], cacc[j]);
    }
  }

  // ---- finalize ------------------------------------------------------------
  const int n = lane & 15, half = lane >> 4;
#pragma unroll
  for (int r = 0; r < 8; ++r) {
    const int m = m0 + r + 8 * half;
    const int s = qt * 64 + m;
    const float invl = 1.f / lrow[r];
#pragma unroll
    for (int j = 0; j < 4; ++j) {
      ctx[((size_t)(b * SS + s) * DD) + h * DH + j * 16 + n] =
          (__bf16)(cacc[j][r] * invl);
    }
  }
}

// ---------------------------------------------------------------------------
// Kernel 3: out_pre = ctx @ Wfc + (q_a + q_s); grid (NTOK/64, D/64), block 128
// ---------------------------------------------------------------------------
__global__ __launch_bounds__(128) void fc_kernel(
    const __bf16* __restrict__ ctx, const __bf16* __restrict__ Wfc,
    const float* __restrict__ q_a, const float* __restrict__ q_s,
    float* __restrict__ out_pre) {
  const int row0 = blockIdx.x * 64;
  const int col0 = blockIdx.y * 64;
  const int tid  = threadIdx.x;
  const int wave = tid >> 5, lane = tid & 31;

  __shared__ __bf16 As[4 * 512];

  v8f acc[4];
#pragma unroll
  for (int j = 0; j < 4; ++j) acc[j] = vzero8();

  for (int kt = 0; kt < 32; ++kt) {
    __syncthreads();
#pragma unroll
    for (int i = 0; i < 2; ++i) {
      const int g = tid + i * 128;
      const int r = g >> 2, c = g & 3;
      const v8bf v = *reinterpret_cast<const v8bf*>(
          &ctx[(size_t)(row0 + r) * DD + kt * 32 + c * 8]);
      *reinterpret_cast<v8bf*>(
          &As[(r >> 4) * 512 + ((r & 15) + 16 * (c & 1)) * 16 + (c >> 1) * 8]) = v;
    }
    __syncthreads();
    const __bf16* wb = &Wfc[((size_t)kt * 64 + (col0 >> 4)) * 512];
    const v16bf b0 = frag_ld(wb + 0 * 512, lane);
    const v16bf b1 = frag_ld(wb + 1 * 512, lane);
    const v16bf b2 = frag_ld(wb + 2 * 512, lane);
    const v16bf b3 = frag_ld(wb + 3 * 512, lane);
    const v16bf a  = frag_ld(&As[wave * 512], lane);
    acc[0] = wmma_bf16(a, b0, acc[0]);
    acc[1] = wmma_bf16(a, b1, acc[1]);
    acc[2] = wmma_bf16(a, b2, acc[2]);
    acc[3] = wmma_bf16(a, b3, acc[3]);
  }

  const int n = lane & 15, half = lane >> 4;
#pragma unroll
  for (int r = 0; r < 8; ++r) {
    const int t = row0 + wave * 16 + r + 8 * half;
#pragma unroll
    for (int j = 0; j < 4; ++j) {
      const size_t idx = (size_t)t * DD + col0 + j * 16 + n;
      out_pre[idx] = acc[j][r] + q_a[idx] + q_s[idx];
    }
  }
}

// ---------------------------------------------------------------------------
// Kernel 4: LayerNorm over last dim (1024)
// ---------------------------------------------------------------------------
__global__ __launch_bounds__(256) void ln_kernel(const float* __restrict__ x,
                                                 const float* __restrict__ gamma,
                                                 const float* __restrict__ beta,
                                                 float* __restrict__ out) {
  const int row = blockIdx.x;
  const float* xr = x + (size_t)row * DD;
  float s = 0.f, ss = 0.f;
#pragma unroll
  for (int i = 0; i < 4; ++i) {
    const float v = xr[threadIdx.x + i * 256];
    s += v;
    ss += v * v;
  }
  for (int d = 1; d < 32; d <<= 1) {
    s  += __shfl_xor(s, d, 32);
    ss += __shfl_xor(ss, d, 32);
  }
  __shared__ float red0[8], red1[8];
  const int wave = threadIdx.x >> 5, lane = threadIdx.x & 31;
  if (lane == 0) { red0[wave] = s; red1[wave] = ss; }
  __syncthreads();
  s = 0.f; ss = 0.f;
#pragma unroll
  for (int w = 0; w < 8; ++w) { s += red0[w]; ss += red1[w]; }
  const float mu   = s * (1.f / DD);
  const float var  = ss * (1.f / DD) - mu * mu;
  const float rstd = rsqrtf(var + 1e-6f);
  float* orow = out + (size_t)row * DD;
#pragma unroll
  for (int i = 0; i < 4; ++i) {
    const int c = threadIdx.x + i * 256;
    orow[c] = (xr[c] - mu) * rstd * gamma[c] + beta[c];
  }
}

// ---------------------------------------------------------------------------
// Launcher
// ---------------------------------------------------------------------------
extern "C" void kernel_launch(void* const* d_in, const int* in_sizes, int n_in,
                              void* d_out, int out_size, void* d_ws, size_t ws_size,
                              hipStream_t stream) {
  (void)in_sizes; (void)n_in; (void)out_size; (void)ws_size;
  const float* q_a = (const float*)d_in[0];
  const float* k_a = (const float*)d_in[1];
  const float* v_a = (const float*)d_in[2];
  const float* q_s = (const float*)d_in[3];
  const float* k_s = (const float*)d_in[4];
  const float* v_s = (const float*)d_in[5];
  const float* Wsrc[7] = {
      (const float*)d_in[6],  (const float*)d_in[7],  (const float*)d_in[8],
      (const float*)d_in[9],  (const float*)d_in[10], (const float*)d_in[11],
      (const float*)d_in[12],
  };
  const float* gamma = (const float*)d_in[13];
  const float* beta  = (const float*)d_in[14];

  char*  ws  = (char*)d_ws;
  size_t off = 0;
  __bf16* Wb[7];
  for (int i = 0; i < 7; ++i) { Wb[i] = (__bf16*)(ws + off); off += (size_t)DD * DD * 2; }
  const size_t qkv_bytes = (size_t)BB * HH * SS * DH * 2;
  __bf16* Qw  = (__bf16*)(ws + off); off += qkv_bytes;
  __bf16* Kw  = (__bf16*)(ws + off); off += qkv_bytes;
  __bf16* Vw  = (__bf16*)(ws + off); off += qkv_bytes;
  __bf16* Ctx = (__bf16*)(ws + off); off += (size_t)NTOK * DD * 2;
  float*  Pre = (float*)(ws + off);  off += (size_t)NTOK * DD * 4;

  for (int i = 0; i < 7; ++i)
    cvt_swizzle_kernel<<<512, 256, 0, stream>>>(Wsrc[i], Wb[i]);

  qkv_proj_kernel<<<dim3(NTOK / 64, HH, 3), 128, 0, stream>>>(
      q_a, q_s, k_a, k_s, v_a, v_s,
      Wb[0], Wb[3], Wb[1], Wb[4], Wb[2], Wb[5], Qw, Kw, Vw);

  attn_kernel<<<dim3(SS / 64, HH, BB), 128, 0, stream>>>(Qw, Kw, Vw, Ctx);

  fc_kernel<<<dim3(NTOK / 64, DD / 64), 128, 0, stream>>>(Ctx, Wb[6], q_a, q_s, Pre);

  ln_kernel<<<NTOK, 256, 0, stream>>>(Pre, gamma, beta, (float*)d_out);
}
